// GraphAttentionLayer2_57698590654936
// MI455X (gfx1250) — matrix operand order repaired and checked by reading, not compiled, not run on hardware
//
#include <hip/hip_runtime.h>
#include <float.h>
#include <stdint.h>

// ---------------- problem constants ----------------
#define GN       10000               // nodes
#define FIN      512
#define FOUT     256
#define NEG_BIG  (-9.0e15f)          // reference mask value
#define TILE     64                  // column tile (two 32-wide WMMA K steps)
#define LROW     72                  // LDS row pitch in bf16 elems (144B: 16B aligned, bank-spread)
#define NWAVES   8                   // waves per workgroup (256 threads, wave32)
#define NTILES   (FOUT / 16)         // 16 feature tiles of 16
#define NJT      ((GN + TILE - 1) / TILE)   // 157 column tiles
#define NSTRIDE  (NJT * TILE)               // 10048 padded node count
#define ROWTILES (GN / 16)                  // 625 row tiles (exact)

typedef __bf16   bf16_t;
typedef bf16_t   v16bf __attribute__((ext_vector_type(16)));
typedef float    v8f   __attribute__((ext_vector_type(8)));
typedef unsigned u32x4 __attribute__((ext_vector_type(4)));
typedef float    f32x4 __attribute__((ext_vector_type(4)));

union ABfrag { v16bf v; u32x4 q[2]; };   // 32 bytes: one 16x32 bf16 A/B operand per lane

#if defined(__has_builtin)
#if __has_builtin(__builtin_amdgcn_tensor_load_to_lds) && __has_builtin(__builtin_amdgcn_s_wait_tensorcnt)
#define HAVE_TDM 1
#endif
#endif

#if defined(HAVE_TDM)
typedef int i32x8v __attribute__((ext_vector_type(8)));
typedef int i32x4v __attribute__((ext_vector_type(4)));

// low 32 bits of a flat pointer into LDS == LDS byte offset (LDS aperture encoding)
__device__ __forceinline__ unsigned lds_offset(const void* p) {
  return (unsigned)(unsigned long long)p;
}

// TDM: DMA one h^T tile (256 rows x 128B contiguous in global) into LDS.
// pad_enable inserts 16B after every 128B -> LDS pitch 144B (= LROW*2), matching
// the bank-conflict-free layout the ds_load_b128 B-fragment loads expect.
__device__ __forceinline__ void tdm_load_hbt_tile(const unsigned short* gsrc, unsigned lds_byte_off) {
  unsigned long long ga = (unsigned long long)gsrc;
  u32x4 g0 = { 1u,                                  // count=1 (one valid descriptor)
               lds_byte_off,                        // lds_addr
               (unsigned)ga,                        // global_addr[31:0]
               (unsigned)((ga >> 32) & 0x01FFFFFFull) | 0x80000000u }; // addr[56:32] | type=2
  i32x8v g1 = { 0x07110000,      // data_size=2B, pad_enable=1, pad_interval=4(128B), pad_amount=3(16B)
                0x40000000,      // tensor_dim0 = 16384 (low16 in bits[31:16])
                0x00010000,      // tensor_dim0 hi=0, tensor_dim1 = 1
                0x40000000,      // tensor_dim1 hi=0, tile_dim0 = 16384
                0,               // tile_dim1 = 0 (unused), tile_dim2 = 0
                16384,           // tensor_dim0_stride low32
                0, 0 };
  i32x4v gz4 = { 0, 0, 0, 0 };
#if __clang_major__ >= 23
  i32x8v gz8 = { 0, 0, 0, 0, 0, 0, 0, 0 };
  __builtin_amdgcn_tensor_load_to_lds(g0, g1, gz4, gz4, gz8, 0);
#else
  __builtin_amdgcn_tensor_load_to_lds(g0, g1, gz4, gz4, 0);
#endif
}
#endif // HAVE_TDM

// pack two f32 -> packed bf16 pair with one v_perm_b32 (+ cheap rounding bias)
__device__ __forceinline__ unsigned pack_bf16(float a, float b) {
  unsigned ua = __float_as_uint(a) + 0x8000u;
  unsigned ub = __float_as_uint(b) + 0x8000u;
  return __builtin_amdgcn_perm(ub, ua, 0x07060302u);  // {hi16(ub), hi16(ua)}
}

// A fragment (16x32 bf16): lane lm = row M, elems e: K=32s+16h+e
__device__ __forceinline__ ABfrag load_afrag(const unsigned short* prow, int s, int h) {
  ABfrag f;
  const u32x4* p = (const u32x4*)(prow + 32 * s + 16 * h);
  f.q[0] = p[0];
  f.q[1] = p[1];        // +8 elems
  return f;
}

// B fragment (32x16 bf16): lane lm = col N, regs r: K = 32s + 2r + 16h
__device__ __forceinline__ ABfrag load_bfrag(const unsigned short* frow, int s, int h) {
  ABfrag f;
  const u32x4* p = (const u32x4*)(frow + 32 * s + 16 * h);
  f.q[0] = p[0];
  f.q[1] = p[1];
  return f;
}

// ================= Stage A: h = bf16(x) @ bf16(W); s1 = h.a1; s2 = h.a2; h -> tiled bf16 =================
__global__ __launch_bounds__(NWAVES * 32)
void gat2_stage_a(const float* __restrict__ x, const float* __restrict__ Wm,
                  const float* __restrict__ av,
                  unsigned short* __restrict__ hbT,
                  float* __restrict__ s1g, float* __restrict__ s2g)
{
  __shared__ unsigned short lds_B[FOUT][LROW];          // W^T k-tile: [feat][64]
  __shared__ unsigned short lds_P[NWAVES][16][LROW];    // x bf16 staging per wave

  const int tid = threadIdx.x;
  const int w = tid >> 5, lane = tid & 31, lm = lane & 15, h = lane >> 4;

  int rt = blockIdx.x * NWAVES + w;
  if (rt >= ROWTILES) rt = ROWTILES - 1;                // tail waves duplicate last tile (same values)
  const int rowbase = rt << 4;
  const int row = rowbase + lm;
  const float* xrow = x + (long long)row * FIN;

  v8f acc[NTILES];
  #pragma unroll
  for (int n = 0; n < NTILES; ++n)
    #pragma unroll
    for (int r = 0; r < 8; ++r) acc[n][r] = 0.0f;

  for (int k0 = 0; k0 < FIN; k0 += TILE) {
    __syncthreads();
    { // cooperative W^T tile: lds_B[feat=tid][kk] = bf16(W[k0+kk][tid]); coalesced across tid
      const float* wsrc = Wm + (long long)k0 * FOUT + tid;
      #pragma unroll
      for (int kk = 0; kk < TILE; kk += 2) {
        float w0 = wsrc[kk * FOUT];
        float w1 = wsrc[(kk + 1) * FOUT];
        *(unsigned*)(&lds_B[tid][kk]) = pack_bf16(w0, w1);
      }
    }
    { // this wave's x tile -> bf16 -> P staging (A-fragment friendly layout)
      const f32x4* xp = (const f32x4*)(xrow + k0 + 32 * h);
      u32x4* pdst = (u32x4*)(&lds_P[w][lm][32 * h]);
      #pragma unroll
      for (int q = 0; q < 4; ++q) {
        f32x4 v0 = xp[2 * q];
        f32x4 v1 = xp[2 * q + 1];
        u32x4 u = { pack_bf16(v0.x, v0.y), pack_bf16(v0.z, v0.w),
                    pack_bf16(v1.x, v1.y), pack_bf16(v1.z, v1.w) };
        pdst[q] = u;
      }
    }
    __syncthreads();
    #pragma unroll
    for (int s = 0; s < 2; ++s) {
      ABfrag A = load_afrag(&lds_P[w][lm][0], s, h);
      #pragma unroll
      for (int n = 0; n < NTILES; ++n) {
        ABfrag B = load_bfrag(&lds_B[16 * n + lm][0], s, h);
        acc[n] = __builtin_amdgcn_wmma_f32_16x16x32_bf16(
            false, A.v, false, B.v, (short)0, acc[n], false, false);
      }
    }
  }

  // s1/s2: per-lane partial dot over this lane's 16 feats, then xor-shuffle reduce over half-wave
  float p1[8], p2[8];
  #pragma unroll
  for (int r = 0; r < 8; ++r) { p1[r] = 0.0f; p2[r] = 0.0f; }
  #pragma unroll
  for (int n = 0; n < NTILES; ++n) {
    float a1 = av[16 * n + lm];
    float a2 = av[FOUT + 16 * n + lm];
    #pragma unroll
    for (int r = 0; r < 8; ++r) {
      p1[r] = fmaf(acc[n][r], a1, p1[r]);
      p2[r] = fmaf(acc[n][r], a2, p2[r]);
    }
  }
  #pragma unroll
  for (int off = 1; off < 16; off <<= 1) {
    #pragma unroll
    for (int r = 0; r < 8; ++r) {
      p1[r] += __shfl_xor(p1[r], off, 32);
      p2[r] += __shfl_xor(p2[r], off, 32);
    }
  }
  if (lm == 0) {
    #pragma unroll
    for (int r = 0; r < 8; ++r) {
      s1g[rowbase + 8 * h + r] = p1[r];
      s2g[rowbase + 8 * h + r] = p2[r];
    }
  }

  // store h as bf16 in tiled layout [node/64][feat][node%64] -> stage B tile fetch is contiguous
  {
    const int jt = rowbase >> 6;
    const int jo = (rowbase & 63) + 8 * h;
    #pragma unroll
    for (int n = 0; n < NTILES; ++n) {
      u32x4 u = { pack_bf16(acc[n][0], acc[n][1]), pack_bf16(acc[n][2], acc[n][3]),
                  pack_bf16(acc[n][4], acc[n][5]), pack_bf16(acc[n][6], acc[n][7]) };
      *(u32x4*)(hbT + (long long)jt * (FOUT * TILE) + (16 * n + lm) * TILE + jo) = u;
    }
  }
}

// ================= Stage B: fused mask + online softmax + attention@h + ELU =================
__global__ __launch_bounds__(NWAVES * 32)
void gat2_stage_b(const int* __restrict__ adj, const unsigned short* __restrict__ hbT,
                  const float* __restrict__ s1g, const float* __restrict__ s2g,
                  float* __restrict__ out)
{
  __shared__ unsigned short lds_Bd[2][FOUT][LROW];      // double-buffered h^T tile (TDM target)
  __shared__ unsigned short lds_P[NWAVES][16][LROW];    // attention-weight staging per wave

  const int tid = threadIdx.x;
  const int w = tid >> 5, lane = tid & 31, lm = lane & 15, h = lane >> 4;

  int rt = blockIdx.x * NWAVES + w;
  if (rt >= ROWTILES) rt = ROWTILES - 1;
  const int rowbase = rt << 4;
  const int row = rowbase + lm;

  const float s1v = s1g[row];
  const int* adjrow = adj + (long long)row * GN;

  v8f acc[NTILES];                                      // 16x256 fp32 accumulator (128 VGPRs)
  #pragma unroll
  for (int n = 0; n < NTILES; ++n)
    #pragma unroll
    for (int r = 0; r < 8; ++r) acc[n][r] = 0.0f;

  float mrun = -3.0e38f;                                // running max (replicated in both halves)
  float lsum = 0.0f;                                    // running softmax denominator

#if defined(HAVE_TDM)
  if (w == 0) tdm_load_hbt_tile(hbT, lds_offset(&lds_Bd[0][0][0]));   // prologue: tile 0 in flight
#endif

  for (int t = 0; t < NJT; ++t) {
    const int buf = t & 1;
    const int j0 = t * TILE;
    __syncthreads();                                    // prior iteration's LDS reads complete

#if defined(HAVE_TDM)
    // kick DMA of the *next* h^T tile; it overlaps the whole softmax phase below
    if (w == 0 && t + 1 < NJT)
      tdm_load_hbt_tile(hbT + (long long)(t + 1) * (FOUT * TILE),
                        lds_offset(&lds_Bd[(t + 1) & 1][0][0]));
#else
    { // fallback: cooperative fill, thread tid copies feat=tid's first 32 nodes (64B contiguous)
      const u32x4* src = (const u32x4*)(hbT + (long long)t * (FOUT * TILE) + tid * TILE);
      u32x4* dst = (u32x4*)(&lds_Bd[buf][tid][0]);
      #pragma unroll
      for (int q = 0; q < 4; ++q) dst[q] = src[q];
    }
#endif

    const int cbase = j0 + 32 * h;                      // this lane: row lm, cols cbase..cbase+31
    if (t + 1 < NJT) __builtin_prefetch(adjrow + cbase + TILE, 0, 1);  // global_prefetch next adj tile

    float sc[32];
    if (j0 + TILE <= GN) {
      const u32x4* ap  = (const u32x4*)(adjrow + cbase);
      const f32x4* s2p = (const f32x4*)(s2g + cbase);
      #pragma unroll
      for (int q = 0; q < 8; ++q) {
        u32x4 a4 = ap[q];
        f32x4 s4 = s2p[q];
        float e0 = s1v + s4.x; e0 = fmaxf(e0, 0.01f * e0);
        float e1 = s1v + s4.y; e1 = fmaxf(e1, 0.01f * e1);
        float e2 = s1v + s4.z; e2 = fmaxf(e2, 0.01f * e2);
        float e3 = s1v + s4.w; e3 = fmaxf(e3, 0.01f * e3);
        sc[4 * q + 0] = ((int)a4.x > 0) ? e0 : NEG_BIG;
        sc[4 * q + 1] = ((int)a4.y > 0) ? e1 : NEG_BIG;
        sc[4 * q + 2] = ((int)a4.z > 0) ? e2 : NEG_BIG;
        sc[4 * q + 3] = ((int)a4.w > 0) ? e3 : NEG_BIG;
      }
    } else {                                            // single partial tail tile
      #pragma unroll
      for (int k = 0; k < 32; ++k) {
        int c = cbase + k;
        bool vld = (c < GN);
        int av2  = vld ? adjrow[c] : 0;
        float sv = vld ? s2g[c] : 0.0f;
        float e = s1v + sv;
        e = fmaxf(e, 0.01f * e);
        sc[k] = vld ? ((av2 > 0) ? e : NEG_BIG) : -3.0e38f;   // padding cols contribute exactly 0
      }
    }

    // per-row max over 64 cols: local 32 + partner half-wave
    float mt = -3.0e38f;
    #pragma unroll
    for (int k = 0; k < 32; ++k) mt = fmaxf(mt, sc[k]);
    mt = fmaxf(mt, __shfl_xor(mt, 16, 32));
    const float m2 = fmaxf(mrun, mt);
    const float scale = __expf(mrun - m2);

    float psum = 0.0f;
    unsigned pk[16];
    #pragma unroll
    for (int k = 0; k < 32; k += 2) {
      float p0 = __expf(sc[k] - m2);
      float p1 = __expf(sc[k + 1] - m2);
      psum += p0 + p1;
      pk[k >> 1] = pack_bf16(p0, p1);
    }
    psum += __shfl_xor(psum, 16, 32);
    lsum = lsum * scale + psum;

    { // stage attention weights (bf16) in the A-fragment layout
      u32x4* pdst = (u32x4*)(&lds_P[w][lm][32 * h]);
      #pragma unroll
      for (int q = 0; q < 4; ++q) {
        u32x4 u = { pk[4 * q + 0], pk[4 * q + 1], pk[4 * q + 2], pk[4 * q + 3] };
        pdst[q] = u;
      }
    }

    if (__ballot(m2 > mrun) != 0ull) {                  // rescale only when running max moved
      float s8[8];
      #pragma unroll
      for (int r = 0; r < 8; ++r) s8[r] = __shfl(scale, r + 8 * h, 32);
      #pragma unroll
      for (int n = 0; n < NTILES; ++n)
        #pragma unroll
        for (int r = 0; r < 8; ++r) acc[n][r] *= s8[r];
    }
    mrun = m2;

#if defined(HAVE_TDM)
    // TDM is in-order per wave: <=1 outstanding guarantees tile t has landed
    if (w == 0) {
      if (t + 1 < NJT) __builtin_amdgcn_s_wait_tensorcnt(1);
      else             __builtin_amdgcn_s_wait_tensorcnt(0);
    }
#endif
    __syncthreads();

    // P(16x64) @ h(64x256): 2 K-steps x 16 feature tiles of v_wmma_f32_16x16x32_bf16
    #pragma unroll
    for (int s = 0; s < 2; ++s) {
      ABfrag A = load_afrag(&lds_P[w][lm][0], s, h);
      #pragma unroll
      for (int n = 0; n < NTILES; ++n) {
        ABfrag B = load_bfrag(&lds_Bd[buf][16 * n + lm][0], s, h);
        acc[n] = __builtin_amdgcn_wmma_f32_16x16x32_bf16(
            false, A.v, false, B.v, (short)0, acc[n], false, false);
      }
    }
  }

  // epilogue: softmax normalize + ELU + store (C layout: M = r + 8h, N = lm)
  const float linv = 1.0f / lsum;
  float l8[8];
  #pragma unroll
  for (int r = 0; r < 8; ++r) l8[r] = __shfl(linv, r + 8 * h, 32);
  #pragma unroll
  for (int n = 0; n < NTILES; ++n) {
    #pragma unroll
    for (int r = 0; r < 8; ++r) {
      float v = acc[n][r] * l8[r];
      v = (v > 0.0f) ? v : (__expf(v) - 1.0f);
      out[(long long)(rowbase + 8 * h + r) * FOUT + 16 * n + lm] = v;
    }
  }
}

extern "C" void kernel_launch(void* const* d_in, const int* in_sizes, int n_in,
                              void* d_out, int out_size, void* d_ws, size_t ws_size,
                              hipStream_t stream) {
  (void)in_sizes; (void)n_in; (void)out_size; (void)ws_size;
  const float* x   = (const float*)d_in[0];
  const int*   adj = (const int*)d_in[1];
  const float* Wm  = (const float*)d_in[2];
  const float* av  = (const float*)d_in[3];
  float* out = (float*)d_out;

  // workspace layout (needs ~5.25 MB):
  //   hbT : bf16 tiled [157][256][64]  = 5,144,576 B
  //   s1  : f32 [10048], s2 : f32 [10048]
  char* ws = (char*)d_ws;
  unsigned short* hbT = (unsigned short*)ws;
  float* s1 = (float*)(ws + (size_t)NJT * FOUT * TILE * 2);
  float* s2 = s1 + NSTRIDE;

  const int blocks = (ROWTILES + NWAVES - 1) / NWAVES;   // 79 workgroups of 8 waves
  gat2_stage_a<<<blocks, NWAVES * 32, 0, stream>>>(x, Wm, av, hbT, s1, s2);
  gat2_stage_b<<<blocks, NWAVES * 32, 0, stream>>>(adj, hbT, s1, s2, out);
}
